// MSAModule_91207925498260
// MI455X (gfx1250) — matrix operand order repaired and checked by reading, not compiled, not run on hardware
//
#include <hip/hip_runtime.h>
#include <math.h>

// ---------------------------------------------------------------------------
// CDNA5 (gfx1250) WMMA types
// ---------------------------------------------------------------------------
typedef __attribute__((ext_vector_type(16))) _Float16 v16h;
typedef __attribute__((ext_vector_type(8)))  _Float16 v8h;
typedef __attribute__((ext_vector_type(4)))  _Float16 v4h;
typedef __attribute__((ext_vector_type(8)))  float    v8f;

#define ACCF  1   // C += result
#define BIASF 2   // add bias[n]
#define SCATF 4   // OPM scatter epilogue: (m=i*32+c, n=j*32+d) -> [(i*192+j)*1024 + c*32 + d]

// ---------------------------------------------------------------------------
// Generic WMMA GEMM: C = alpha * (A @ W) (+bias) (+C)
//   A: [M,K] f32 row major (transA -> stored [K,M], leading dim lda)
//   W: [K,N] f32 row major
// Contract (guaranteed by all call sites): M % 32 == 0, K % 32 == 0.
// Only N may be ragged (N=4 bias projection) -> uniform-branch guarded path.
// One workgroup = 4 waves = 32x32 C tile; k-loop in steps of 32 via LDS f16.
// A staged row-major [m][k]; W staged transposed [n][k] so both WMMA
// fragments are contiguous 16B LDS chunks (ds_load_b128).
// ---------------------------------------------------------------------------
__global__ __launch_bounds__(128) void k_gemm(
    const float* __restrict__ A, int lda, int transA,
    const float* __restrict__ W, int ldw,
    const float* __restrict__ bias,
    float* __restrict__ C, int ldc,
    int M, int N, int K, float alpha, int flags)
{
  __shared__ _Float16 As[32][40];   // [m][k], 80B row stride (16B aligned)
  __shared__ _Float16 Wt[32][40];   // [n][k]
  const int tid  = threadIdx.x;
  const int lane = tid & 31;
  const int wave = tid >> 5;
  const int m0   = blockIdx.x * 32;
  const int n0   = blockIdx.y * 32;
  const int wm   = (wave >> 1) << 4;
  const int wn   = (wave & 1) << 4;
  const int mrow = lane & 15;
  const int half = lane >> 4;
  const bool fullN = ((N & 31) == 0);   // uniform across the grid

  v8f acc = {};
  for (int k0 = 0; k0 < K; k0 += 32) {
    // ---- stage A tile (32 m x 32 k) ----
    if (!transA) {
      for (int f = tid; f < 256; f += 128) {            // 256 float4 = 1024 elems
        int r = f >> 3, c4 = (f & 7) << 2;
        const float4 v = *(const float4*)(A + (size_t)(m0 + r) * lda + (k0 + c4));
        v4h hv = { (_Float16)v.x, (_Float16)v.y, (_Float16)v.z, (_Float16)v.w };
        *(v4h*)&As[r][c4] = hv;
      }
    } else {
      for (int e = tid; e < 1024; e += 128) {           // lanes sweep m (coalesced)
        int r = e & 31, c = e >> 5;                     // r = m, c = k
        As[r][c] = (_Float16)A[(size_t)(k0 + c) * lda + (m0 + r)];
      }
    }
    // ---- stage W tile transposed -> Wt[n][k] ----
    if (fullN) {
      for (int f = tid; f < 256; f += 128) {
        int r = f >> 3, c4 = (f & 7) << 2;              // r = k, c4 = n base
        const float4 v = *(const float4*)(W + (size_t)(k0 + r) * ldw + (n0 + c4));
        Wt[c4 + 0][r] = (_Float16)v.x;
        Wt[c4 + 1][r] = (_Float16)v.y;
        Wt[c4 + 2][r] = (_Float16)v.z;
        Wt[c4 + 3][r] = (_Float16)v.w;
      }
    } else {
      for (int e = tid; e < 1024; e += 128) {
        int r = e >> 5, c = e & 31;                     // r = k, c = n
        int nn = n0 + c;
        float v = (nn < N) ? W[(size_t)(k0 + r) * ldw + nn] : 0.f;
        Wt[c][r] = (_Float16)v;
      }
    }
    __syncthreads();
    // ---- fragments: contiguous 16B LDS chunks ----
    const _Float16* arow = &As[wm + mrow][0];
    v8h a0 = *(const v8h*)(arow + (half << 3));          // k = 8*half + 0..7
    v8h a1 = *(const v8h*)(arow + 16 + (half << 3));     // k = 16 + 8*half + 0..7
    v16h af = __builtin_shufflevector(a0, a1, 0, 1, 2, 3, 4, 5, 6, 7,
                                      8, 9, 10, 11, 12, 13, 14, 15);
    const _Float16* brow = &Wt[wn + mrow][0];
    v8h b0 = *(const v8h*)(brow + (half << 3));          // k = 8*half + 0..7
    v8h b1 = *(const v8h*)(brow + 16 + (half << 3));     // k = 16 + 8*half + 0..7
    v16h bf = __builtin_shufflevector(b0, b1, 0, 1, 2, 3, 4, 5, 6, 7,
                                      8, 9, 10, 11, 12, 13, 14, 15);
    acc = __builtin_amdgcn_wmma_f32_16x16x32_f16(false, af, false, bf,
                                                 (short)0, acc, false, false);
    __syncthreads();
  }

  const int n  = n0 + wn + mrow;
  const int mb = m0 + wm + (half << 3);
#pragma unroll
  for (int r = 0; r < 8; ++r) {
    int mm = mb + r;
    if (n < N) {
      float v = alpha * acc[r];
      if (flags & BIASF) v += bias[n];
      size_t off;
      if (flags & SCATF) {
        int i = mm >> 5, c = mm & 31, j = n >> 5, d = n & 31;
        off = ((size_t)(i * 192 + j) << 10) + (size_t)((c << 5) + d);
      } else {
        off = (size_t)mm * ldc + n;
      }
      if (flags & ACCF) C[off] += v; else C[off] = v;
    }
  }
}

// ---------------------------------------------------------------------------
// Row LayerNorm, blockDim == C (64 or 128). T!=0: read input transposed
// over leading (T x T) dims: row r -> (i=r/T, j=r%T), read x[j,i,:].
// ---------------------------------------------------------------------------
__global__ void k_ln(const float* __restrict__ x, float* __restrict__ y,
                     const float* __restrict__ g, const float* __restrict__ b,
                     int C, int T)
{
  int row = blockIdx.x;
  int c = threadIdx.x;
  size_t inoff;
  if (T) { int i = row / T, j = row - i * T; inoff = ((size_t)(j * T + i)) * C + c; }
  else   { inoff = (size_t)row * C + c; }
  float v = x[inoff];
  __shared__ float red[128];
  red[c] = v; __syncthreads();
  for (int s = C >> 1; s > 0; s >>= 1) { if (c < s) red[c] += red[c + s]; __syncthreads(); }
  float mean = red[0] * (1.f / C);
  __syncthreads();
  float d = v - mean;
  red[c] = d * d; __syncthreads();
  for (int s = C >> 1; s > 0; s >>= 1) { if (c < s) red[c] += red[c + s]; __syncthreads(); }
  float var = red[0] * (1.f / C);
  y[(size_t)row * C + c] = d * rsqrtf(var + 1e-5f) * g[c] + b[c];
}

// ---------------------------------------------------------------------------
// Triangle multiplicative einsum:
//   outgoing: X[i,j,c] = sum_k A[i,k,c] * B[j,k,c]
//   incoming: X[i,j,c] = sum_k A[k,i,c] * B[k,j,c]
// grid = (144 ij-tiles, 32 c-groups), block = 128 (wave -> one channel c).
// ---------------------------------------------------------------------------
__global__ __launch_bounds__(128) void k_trimult(
    const float* __restrict__ Ab, const float* __restrict__ Bb,
    float* __restrict__ Out, int outgoing)
{
  const int tile = blockIdx.x;
  const int i0 = (tile / 12) * 16, j0 = (tile % 12) * 16;
  const int wave = threadIdx.x >> 5, lane = threadIdx.x & 31;
  const int c = blockIdx.y * 4 + wave;
  const int mrow = lane & 15, half = lane >> 4;

  v8f acc = {};
  for (int k0 = 0; k0 < 192; k0 += 32) {
    v16h af, bf;
#pragma unroll
    for (int e = 0; e < 16; ++e) {
      int kk = k0 + ((e >> 3) << 4) + (half << 3) + (e & 7);
      int i = i0 + mrow;
      size_t off = outgoing ? ((size_t)(i * 192 + kk) << 7) + c
                            : ((size_t)(kk * 192 + i) << 7) + c;
      af[e] = (_Float16)Ab[off];
    }
#pragma unroll
    for (int e = 0; e < 16; ++e) {
      int kk = k0 + ((e >> 3) << 4) + (half << 3) + (e & 7);
      int j = j0 + mrow;
      size_t off = outgoing ? ((size_t)(j * 192 + kk) << 7) + c
                            : ((size_t)(kk * 192 + j) << 7) + c;
      bf[e] = (_Float16)Bb[off];
    }
    acc = __builtin_amdgcn_wmma_f32_16x16x32_f16(false, af, false, bf,
                                                 (short)0, acc, false, false);
  }
  const int n = j0 + mrow;
  const int mb = i0 + (half << 3);
#pragma unroll
  for (int r = 0; r < 8; ++r)
    Out[((size_t)((mb + r) * 192 + n) << 7) + c] = acc[r];
}

// ---------------------------------------------------------------------------
// Fused triangle attention (starting-node form; ending handled by transposed
// LN + transposed residual add outside). grid = (i=192, h=4, jt=12), 1 wave.
//   logits[j,k] = (1/sqrt(32)) sum_c q[i,j,h,c] k[i,k,h,c] + bw[j,k,h]
//   o[i,j,h,c]  = sigmoid(g) * sum_k softmax_k(logits) v[i,k,h,c]
// ---------------------------------------------------------------------------
__global__ __launch_bounds__(32) void k_triattn(
    const float* __restrict__ q, const float* __restrict__ kb,
    const float* __restrict__ vb, const float* __restrict__ gb,
    const float* __restrict__ bw, float* __restrict__ ob)
{
  __shared__ float    L[16][200];
  __shared__ _Float16 P[16][200];
  const int i = blockIdx.x, h = blockIdx.y, j0 = blockIdx.z * 16;
  const int lane = threadIdx.x;
  const int mrow = lane & 15, half = lane >> 4;
  const int hc = h << 5;

  v16h qf;
#pragma unroll
  for (int e = 0; e < 16; ++e) {
    int cc = ((e >> 3) << 4) + (half << 3) + (e & 7);
    qf[e] = (_Float16)q[((size_t)(i * 192 + j0 + mrow) << 7) + hc + cc];
  }
  for (int kt = 0; kt < 12; ++kt) {
    v16h kf;
#pragma unroll
    for (int e = 0; e < 16; ++e) {
      int cc = ((e >> 3) << 4) + (half << 3) + (e & 7);
      kf[e] = (_Float16)kb[((size_t)(i * 192 + kt * 16 + mrow) << 7) + hc + cc];
    }
    v8f acc = {};
    acc = __builtin_amdgcn_wmma_f32_16x16x32_f16(false, qf, false, kf,
                                                 (short)0, acc, false, false);
#pragma unroll
    for (int r = 0; r < 8; ++r) {
      int j = (half << 3) + r;
      int kk = kt * 16 + mrow;
      L[j][kk] = acc[r] * 0.17677669529663687f
               + bw[((size_t)((j0 + j) * 192 + kk) << 2) + h];
    }
  }
  __syncthreads();
  if (lane < 16) {
    float mx = -1e30f;
    for (int k = 0; k < 192; ++k) mx = fmaxf(mx, L[lane][k]);
    float s = 0.f;
    for (int k = 0; k < 192; ++k) { float e = expf(L[lane][k] - mx); L[lane][k] = e; s += e; }
    float inv = 1.f / s;
    for (int k = 0; k < 192; ++k) P[lane][k] = (_Float16)(L[lane][k] * inv);
  }
  __syncthreads();
  for (int nt = 0; nt < 2; ++nt) {
    v8f o = {};
    for (int k0 = 0; k0 < 192; k0 += 32) {
      v16h pf, vf;
#pragma unroll
      for (int e = 0; e < 16; ++e)
        pf[e] = P[mrow][k0 + ((e >> 3) << 4) + (half << 3) + (e & 7)];
#pragma unroll
      for (int e = 0; e < 16; ++e) {
        int kk = k0 + ((e >> 3) << 4) + (half << 3) + (e & 7);
        vf[e] = (_Float16)vb[((size_t)(i * 192 + kk) << 7) + hc + nt * 16 + mrow];
      }
      o = __builtin_amdgcn_wmma_f32_16x16x32_f16(false, pf, false, vf,
                                                 (short)0, o, false, false);
    }
#pragma unroll
    for (int r = 0; r < 8; ++r) {
      int j = j0 + (half << 3) + r;
      int cc = nt * 16 + mrow;
      size_t off = ((size_t)(i * 192 + j) << 7) + hc + cc;
      float gg = gb[off];
      ob[off] = o[r] / (1.f + expf(-gg));
    }
  }
}

// ---------------------------------------------------------------------------
// Elementwise kernels
// ---------------------------------------------------------------------------
__global__ void k_copy(float* __restrict__ d, const float* __restrict__ s, unsigned n) {
  unsigned i = blockIdx.x * 256 + threadIdx.x;
  if (i < n) d[i] = s[i];
}
__global__ void k_addbc(float* __restrict__ m, const float* __restrict__ s, unsigned n) {
  unsigned i = blockIdx.x * 256 + threadIdx.x;
  if (i < n) {
    unsigned c = i & 63, row = i >> 6, t = row % 192;
    m[i] += s[t * 64 + c];
  }
}
__global__ void k_sigmul(float* __restrict__ o, const float* __restrict__ a,
                         const float* __restrict__ b, unsigned n) {
  unsigned i = blockIdx.x * 256 + threadIdx.x;
  if (i < n) o[i] = b[i] / (1.f + expf(-a[i]));
}
__global__ void k_silumul(float* __restrict__ o, const float* __restrict__ a,
                          const float* __restrict__ b, unsigned n) {
  unsigned i = blockIdx.x * 256 + threadIdx.x;
  if (i < n) { float x = a[i]; o[i] = (x / (1.f + expf(-x))) * b[i]; }
}
__global__ void k_sigmul_add(float* __restrict__ d, const float* __restrict__ g,
                             const float* __restrict__ t, unsigned n) {
  unsigned i = blockIdx.x * 256 + threadIdx.x;
  if (i < n) d[i] += t[i] / (1.f + expf(-g[i]));
}
__global__ void k_addT(float* __restrict__ pair, const float* __restrict__ src, unsigned n) {
  unsigned idx = blockIdx.x * 256 + threadIdx.x;
  if (idx < n) {
    unsigned c = idx & 127, rest = idx >> 7;
    unsigned i = rest / 192, j = rest % 192;
    pair[((size_t)(j * 192 + i) << 7) + c] += src[idx];
  }
}

// ---------------------------------------------------------------------------
// Host orchestration
// ---------------------------------------------------------------------------
static inline dim3 ewg(unsigned n) { return dim3((n + 255u) / 256u); }

static void gemm(hipStream_t s, const float* A, int lda, int tA,
                 const float* W, int ldw, const float* bias,
                 float* C, int ldc, int M, int N, int K, float alpha, int flags)
{
  dim3 g((M + 31) / 32, (N + 31) / 32);
  k_gemm<<<g, 128, 0, s>>>(A, lda, tA, W, ldw, bias, C, ldc, M, N, K, alpha, flags);
}

extern "C" void kernel_launch(void* const* d_in, const int* in_sizes, int n_in,
                              void* d_out, int out_size, void* d_ws, size_t ws_size,
                              hipStream_t stream)
{
  (void)in_sizes; (void)n_in; (void)out_size; (void)ws_size;
  const float* msa    = (const float*)d_in[0];   // [1,512,192,64]
  const float* pin    = (const float*)d_in[1];   // [1,192,192,128]
  const float* single = (const float*)d_in[2];   // [1,192,384]
  auto P = [&](int i) { return (const float*)d_in[i]; };
  // jax pytree (sorted-key) flatten: blocks[b] leaves at 3+b*60, then
  // linear_msa_w=123, linear_single_w=124. Per-block sub-module offsets:
  //   mpa +0..7 (dead), msa_trans +8..12 (dead),
  //   opm +13: a_w,b_w,ln_b,ln_g,o_b,o_w
  //   pair_trans +19: a_w,b_w,ln_b,ln_g,o_w
  //   tae +24 / tas +32: b_w,g_w,k_w,ln_b,ln_g,o_w,q_w,v_w
  //   tmi +40 / tmo +50: g_w,ga_w,gb_w,ln2_b,ln2_g,ln_b,ln_g,o_w,pa_w,pb_w

  float* ws = (float*)d_ws;
  float* m    = ws;                 // 98304*64  = 6291456 f (persistent)
  float* mln  = ws + 6291456;       // 6291456 f
  float* bufA = ws + 12582912;      // 98304*32  = 3145728 f
  float* bufB = ws + 15728640;      // 3145728 f
  float* BIG  = ws + 18874368;      // 6144*6144 = 37748736 f  (8 pair slots)
  const size_t PSLOT = 4718592;     // 192*192*128
  float* S[8];
  for (int k = 0; k < 8; ++k) S[k] = BIG + (size_t)k * PSLOT;
  float* pair = (float*)d_out;

  const unsigned NP = 192 * 192 * 128;   // pair elements
  const unsigned NM = 98304 * 64;        // msa elements

  // ---- Embedding: m = msa@Wm + broadcast(single@Ws)  (dead-code pruned MSA
  //      stream: only this m feeds OPM in every block) ----
  gemm(stream, msa, 64, 0, P(123), 64, nullptr, m, 64, 98304, 64, 64, 1.f, 0);
  gemm(stream, single, 384, 0, P(124), 64, nullptr, bufA, 64, 192, 64, 384, 1.f, 0);
  k_addbc<<<ewg(NM), 256, 0, stream>>>(m, bufA, NM);
  k_copy<<<ewg(NP), 256, 0, stream>>>(pair, pin, NP);

  for (int b = 0; b < 2; ++b) {
    const int B0 = 3 + b * 60;

    // ---- OuterProductMean:  pair += ((1/S) a^T b)@o_w + o_b ----
    k_ln<<<98304, 64, 0, stream>>>(m, mln, P(B0 + 16), P(B0 + 15), 64, 0);
    gemm(stream, mln, 64, 0, P(B0 + 13), 32, nullptr, bufA, 32, 98304, 32, 64, 1.f, 0);
    gemm(stream, mln, 64, 0, P(B0 + 14), 32, nullptr, bufB, 32, 98304, 32, 64, 1.f, 0);
    // E[(i,c),(j,d)] scattered to [i,j,c*32+d], scaled by 1/512 (mean over S)
    gemm(stream, bufA, 6144, 1, bufB, 6144, nullptr, BIG, 0,
         6144, 6144, 512, 1.f / 512.f, SCATF);
    gemm(stream, BIG, 1024, 0, P(B0 + 18), 128, P(B0 + 17), pair, 128,
         36864, 128, 1024, 1.f, ACCF | BIASF);

    // ---- Triangle multiplicative updates (outgoing tmo, incoming tmi) ----
    for (int v = 0; v < 2; ++v) {
      const int tb = B0 + (v == 0 ? 50 : 40);  // tmo then tmi
      const int outgoing = (v == 0) ? 1 : 0;
      k_ln<<<36864, 128, 0, stream>>>(pair, S[0], P(tb + 6), P(tb + 5), 128, 0);
      gemm(stream, S[0], 128, 0, P(tb + 1), 128, nullptr, S[1], 128, 36864, 128, 128, 1.f, 0); // ga
      gemm(stream, S[0], 128, 0, P(tb + 8), 128, nullptr, S[2], 128, 36864, 128, 128, 1.f, 0); // pa
      k_sigmul<<<ewg(NP), 256, 0, stream>>>(S[1], S[1], S[2], NP);                             // a
      gemm(stream, S[0], 128, 0, P(tb + 2), 128, nullptr, S[2], 128, 36864, 128, 128, 1.f, 0); // gb
      gemm(stream, S[0], 128, 0, P(tb + 9), 128, nullptr, S[3], 128, 36864, 128, 128, 1.f, 0); // pb
      k_sigmul<<<ewg(NP), 256, 0, stream>>>(S[2], S[2], S[3], NP);                             // b
      k_trimult<<<dim3(144, 32), 128, 0, stream>>>(S[1], S[2], S[3], outgoing);                // x
      k_ln<<<36864, 128, 0, stream>>>(S[3], S[4], P(tb + 4), P(tb + 3), 128, 0);               // ln2(x)
      gemm(stream, S[0], 128, 0, P(tb + 0), 128, nullptr, S[5], 128, 36864, 128, 128, 1.f, 0); // g
      gemm(stream, S[4], 128, 0, P(tb + 7), 128, nullptr, S[6], 128, 36864, 128, 128, 1.f, 0); // @o_w
      k_sigmul_add<<<ewg(NP), 256, 0, stream>>>(pair, S[5], S[6], NP);
    }

    // ---- Triangle attention (starting tas, ending tae via transpose) ----
    for (int v = 0; v < 2; ++v) {
      const int tb = B0 + (v == 0 ? 32 : 24);  // tas then tae
      const int trans = (v == 0) ? 0 : 192;
      k_ln<<<36864, 128, 0, stream>>>(pair, S[0], P(tb + 4), P(tb + 3), 128, trans);
      gemm(stream, S[0], 128, 0, P(tb + 6), 128, nullptr, S[1], 128, 36864, 128, 128, 1.f, 0); // q
      gemm(stream, S[0], 128, 0, P(tb + 2), 128, nullptr, S[2], 128, 36864, 128, 128, 1.f, 0); // k
      gemm(stream, S[0], 128, 0, P(tb + 7), 128, nullptr, S[3], 128, 36864, 128, 128, 1.f, 0); // v
      gemm(stream, S[0], 128, 0, P(tb + 1), 128, nullptr, S[4], 128, 36864, 128, 128, 1.f, 0); // g
      gemm(stream, S[0], 128, 0, P(tb + 0),   4, nullptr, S[5],   4, 36864,   4, 128, 1.f, 0); // bias
      k_triattn<<<dim3(192, 4, 12), 32, 0, stream>>>(S[1], S[2], S[3], S[4], S[5], S[6]);
      if (!trans) {
        gemm(stream, S[6], 128, 0, P(tb + 5), 128, nullptr, pair, 128,
             36864, 128, 128, 1.f, ACCF);
      } else {
        gemm(stream, S[6], 128, 0, P(tb + 5), 128, nullptr, S[7], 128,
             36864, 128, 128, 1.f, 0);
        k_addT<<<ewg(NP), 256, 0, stream>>>(pair, S[7], NP);
      }
    }

    // ---- Pair transition (row-halved to fit workspace) ----
    {
      const int tb = B0 + 19;
      k_ln<<<36864, 128, 0, stream>>>(pair, S[0], P(tb + 3), P(tb + 2), 128, 0);
      for (int half = 0; half < 2; ++half) {
        const int R0 = half * 18432;
        const unsigned NH = 18432u * 512u;
        gemm(stream, S[0] + (size_t)R0 * 128, 128, 0, P(tb + 0), 512, nullptr,
             S[1], 512, 18432, 512, 128, 1.f, 0);                 // a (spans S1..S2)
        gemm(stream, S[0] + (size_t)R0 * 128, 128, 0, P(tb + 1), 512, nullptr,
             S[3], 512, 18432, 512, 128, 1.f, 0);                 // b (spans S3..S4)
        k_silumul<<<ewg(NH), 256, 0, stream>>>(S[1], S[1], S[3], NH);
        gemm(stream, S[1], 512, 0, P(tb + 4), 128, nullptr,
             pair + (size_t)R0 * 128, 128, 18432, 128, 512, 1.f, ACCF);
      }
    }
  }
}
